// MolEncoder_60773787238887
// MI455X (gfx1250) — matrix operand order repaired and checked by reading, not compiled, not run on hardware
//
#include <hip/hip_runtime.h>
#include <hip/hip_bf16.h>

// ---------------------------------------------------------------------------
// MolEncoder (GATv2 x4 + pooling + MLP + L2norm) for MI455X (gfx1250, wave32)
//   - GEMMs: v_wmma_f32_16x16x32_bf16, 64x64 block tile, 8 waves,
//     2 WMMA / wave / K-chunk (shared A fragment), fully unrolled K loop,
//     clamped (branch-free) tile loads, b128 LDS traffic, global prefetch
//   - Edge softmax: 3-pass atomics, wave-per-edge, float4 loads, shfl reductions
//   - edge_attr @ We recomputed in-register from LDS (saves ~0.65 GB/layer HBM)
// ---------------------------------------------------------------------------

typedef __bf16 bf16_t;
typedef bf16_t v16bf __attribute__((ext_vector_type(16)));
typedef float  v8f   __attribute__((ext_vector_type(8)));

union FragBF { unsigned int u[8]; uint4 q[2]; v16bf v; };

__device__ __forceinline__ unsigned short f2bf(float x) {
    unsigned int u = __float_as_uint(x);
    unsigned int r = u + 0x7FFFu + ((u >> 16) & 1u);   // round-to-nearest-even
    return (unsigned short)(r >> 16);
}
__device__ __forceinline__ unsigned int pk2bf(float lo, float hi) {
    return (unsigned int)f2bf(lo) | ((unsigned int)f2bf(hi) << 16);
}

__device__ __forceinline__ float gelu_exact(float x) {
    return 0.5f * x * (1.0f + erff(x * 0.70710678118654752f));
}

// monotone float<->uint mapping for atomicMax on floats
__device__ __forceinline__ unsigned int ford(float x) {
    unsigned int b = __float_as_uint(x);
    return (b & 0x80000000u) ? ~b : (b | 0x80000000u);
}
__device__ __forceinline__ float funord(unsigned int m) {
    unsigned int b = (m & 0x80000000u) ? (m ^ 0x80000000u) : ~m;
    return __uint_as_float(b);
}
#define NEG_INF_ORD 0x007FFFFFu   // ford(-inf)

// ---------------------------------------------------------------------------
// WMMA GEMM: Out[M,256] = act(A[M,KC] @ W[KC,256] + bias)
// KC is compile-time (161 for projection, 256 elsewhere); Nc fixed at 256.
// Block: 256 threads = 8 waves; 64x64 output tile; wave = 16x32 strip.
// Loads are clamp-addressed (no per-element masks in the hot loop).
// ---------------------------------------------------------------------------
template<int KC>
__global__ __launch_bounds__(256) void gemm_wmma(
    const float* __restrict__ A, const float* __restrict__ W,
    const float* __restrict__ bias, float* __restrict__ Out,
    int M, int act)
{
    constexpr int NC = 256;
    alignas(16) __shared__ unsigned short lA[64][40];  // [m][k], pitch 80B
    alignas(16) __shared__ unsigned short lB[64][40];  // [n][k] (B transposed)

    const int t    = threadIdx.x;
    const int lane = t & 31;
    const int wave = t >> 5;          // 0..7
    const int wr   = wave >> 1;       // 0..3 : row strip
    const int wc   = wave & 1;        // 0..1 : col pair
    const int m0   = blockIdx.y * 64;
    const int n0   = blockIdx.x * 64;
    const int hf   = lane >> 4;
    const int lrow = lane & 15;

    // tile-load mapping (per thread: 8 A elems, 8 B elems per chunk)
    const int am  = t >> 2;               // 0..63 A row in tile
    const int aks = (t & 3) << 3;         // 0,8,16,24
    const int bn  = t >> 2;               // 0..63 B col in tile
    const int bks = (t & 3) << 3;

    // clamp read row so loads need no masks (writes stay guarded)
    const int amr = (m0 + am < M) ? (m0 + am) : (M - 1);
    const float* __restrict__ Arow = A + (long)amr * KC;
    const float* __restrict__ Wcol = W + (n0 + bn);    // n always < NC (grid exact)

    v8f acc0 = {}, acc1 = {};

    auto mma_step = [&]() {
        FragBF fa, fb0, fb1;
        const unsigned short* arow = lA[wr * 16 + lrow];
        fa.q[0] = *(const uint4*)&arow[hf * 16];        // K 0..7 / 16..23
        fa.q[1] = *(const uint4*)&arow[hf * 16 + 8];    // K 8..15 / 24..31
        const unsigned short* br0 = lB[wc * 32 + lrow];
        fb0.q[0] = *(const uint4*)&br0[hf * 16];
        fb0.q[1] = *(const uint4*)&br0[hf * 16 + 8];
        const unsigned short* br1 = lB[wc * 32 + 16 + lrow];
        fb1.q[0] = *(const uint4*)&br1[hf * 16];
        fb1.q[1] = *(const uint4*)&br1[hf * 16 + 8];
        acc0 = __builtin_amdgcn_wmma_f32_16x16x32_bf16(
                   false, fa.v, false, fb0.v, (short)0, acc0, false, false);
        acc1 = __builtin_amdgcn_wmma_f32_16x16x32_bf16(
                   false, fa.v, false, fb1.v, (short)0, acc1, false, false);
    };

    constexpr int KFULL = KC & ~31;

#pragma unroll
    for (int k0 = 0; k0 < KFULL; k0 += 32) {
        // ---- A tile (vector path iff rows are 16B-phased) ---------------
        uint4 wa;
        if constexpr ((KC & 3) == 0) {
            float4 p0 = *(const float4*)&Arow[k0 + aks];
            float4 p1 = *(const float4*)&Arow[k0 + aks + 4];
            wa.x = pk2bf(p0.x, p0.y); wa.y = pk2bf(p0.z, p0.w);
            wa.z = pk2bf(p1.x, p1.y); wa.w = pk2bf(p1.z, p1.w);
        } else {
            float v[8];
#pragma unroll
            for (int j = 0; j < 8; ++j) v[j] = Arow[k0 + aks + j];
            wa.x = pk2bf(v[0], v[1]); wa.y = pk2bf(v[2], v[3]);
            wa.z = pk2bf(v[4], v[5]); wa.w = pk2bf(v[6], v[7]);
        }
        // ---- B tile: column per thread, immediate-offset loads ----------
        float vb[8];
#pragma unroll
        for (int j = 0; j < 8; ++j) vb[j] = Wcol[(long)(k0 + bks + j) * NC];
        uint4 wb;
        wb.x = pk2bf(vb[0], vb[1]); wb.y = pk2bf(vb[2], vb[3]);
        wb.z = pk2bf(vb[4], vb[5]); wb.w = pk2bf(vb[6], vb[7]);

        if (k0 + 32 < KC) {   // compile-time after unroll
            __builtin_prefetch(&Arow[k0 + 32 + aks], 0, 0);
            __builtin_prefetch(&Wcol[(long)(k0 + 32 + bks) * NC], 0, 0);
        }

        *(uint4*)&lA[am][aks] = wa;
        *(uint4*)&lB[bn][bks] = wb;
        __syncthreads();
        mma_step();
        __syncthreads();
    }

    // ---- ragged K tail (only instantiated for KC%32 != 0, i.e. KC=161) ---
    if constexpr ((KC & 31) != 0) {
        float va[8], vb[8];
#pragma unroll
        for (int j = 0; j < 8; ++j) {
            int k  = KFULL + aks + j;
            int kc = (k < KC) ? k : (KC - 1);
            float v = Arow[kc];
            va[j] = (k < KC) ? v : 0.0f;
        }
#pragma unroll
        for (int j = 0; j < 8; ++j) {
            int k  = KFULL + bks + j;
            int kc = (k < KC) ? k : (KC - 1);
            float v = Wcol[(long)kc * NC];
            vb[j] = (k < KC) ? v : 0.0f;
        }
        uint4 wa, wb;
        wa.x = pk2bf(va[0], va[1]); wa.y = pk2bf(va[2], va[3]);
        wa.z = pk2bf(va[4], va[5]); wa.w = pk2bf(va[6], va[7]);
        wb.x = pk2bf(vb[0], vb[1]); wb.y = pk2bf(vb[2], vb[3]);
        wb.z = pk2bf(vb[4], vb[5]); wb.w = pk2bf(vb[6], vb[7]);
        *(uint4*)&lA[am][aks] = wa;
        *(uint4*)&lB[bn][bks] = wb;
        __syncthreads();
        mma_step();
        __syncthreads();
    }

    // ---- writeback: N = lane%16, M = vgpr + 8*half --------------------------
    const int nA = n0 + wc * 32 + lrow;
    const int nB = nA + 16;
    const float biasA = bias[nA];
    const float biasB = bias[nB];
#pragma unroll
    for (int i = 0; i < 8; ++i) {
        int m = m0 + wr * 16 + i + 8 * hf;
        if (m < M) {
            float v0 = acc0[i] + biasA;
            float v1 = acc1[i] + biasB;
            if (act == 1) { v0 = gelu_exact(v0); v1 = gelu_exact(v1); }
            Out[(long)m * NC + nA] = v0;
            Out[(long)m * NC + nB] = v1;
        }
    }
}

// ---------------------------------------------------------------------------
// fill helper
// ---------------------------------------------------------------------------
__global__ void fill_u32(unsigned int* __restrict__ p, unsigned int v, long n) {
    long i = (long)blockIdx.x * blockDim.x + threadIdx.x;
    if (i < n) p[i] = v;
}

// ---------------------------------------------------------------------------
// Pass 1: per-edge logits + segment max (wave per edge, 8 feats/lane)
// e = edge_attr @ We computed in-register from LDS-cached We (11x256).
// ---------------------------------------------------------------------------
__global__ __launch_bounds__(256) void edge_logits(
    const float* __restrict__ xl, const float* __restrict__ xr,
    const float* __restrict__ eattr, const int* __restrict__ src,
    const int* __restrict__ dst, const float* __restrict__ We,
    const float* __restrict__ att, float* __restrict__ exbuf,
    unsigned int* __restrict__ nodemax, int E)
{
    __shared__ float sWe[11][256];
    __shared__ float sAtt[256];
    for (int i = threadIdx.x; i < 11 * 256; i += 256) sWe[i / 256][i % 256] = We[i];
    sAtt[threadIdx.x] = att[threadIdx.x];
    __syncthreads();

    const int lane = threadIdx.x & 31;
    const int wave = threadIdx.x >> 5;
    long e = (long)blockIdx.x * 8 + wave;
    if (e >= E) return;

    const int s = src[e], d = dst[e];
    const int f = lane * 8;            // 8 contiguous features; h = f/64 = lane/8

    float ea[11];
#pragma unroll
    for (int k = 0; k < 11; ++k) ea[k] = eattr[e * 11 + k];

    float4 a0 = *(const float4*)&xl[(long)s * 256 + f];
    float4 a1 = *(const float4*)&xl[(long)s * 256 + f + 4];
    float4 b0 = *(const float4*)&xr[(long)d * 256 + f];
    float4 b1 = *(const float4*)&xr[(long)d * 256 + f + 4];
    float va[8] = { a0.x + b0.x, a0.y + b0.y, a0.z + b0.z, a0.w + b0.w,
                    a1.x + b1.x, a1.y + b1.y, a1.z + b1.z, a1.w + b1.w };

    float part = 0.0f;
#pragma unroll
    for (int j = 0; j < 8; ++j) {
        float m = va[j];
#pragma unroll
        for (int k = 0; k < 11; ++k) m += ea[k] * sWe[k][f + j];
        float sv = (m > 0.0f) ? m : 0.2f * m;      // leaky_relu(0.2)
        part += sv * sAtt[f + j];
    }
    // reduce 8 lanes sharing one head
    for (int o = 1; o < 8; o <<= 1) part += __shfl_xor(part, o, 32);

    if ((lane & 7) == 0) {
        int h = lane >> 3;
        exbuf[e * 4 + h] = part;                   // stash raw logit
        atomicMax(&nodemax[(long)d * 4 + h], ford(part));
    }
}

// ---------------------------------------------------------------------------
// Pass 2: ex = exp(logit - max), segment-sum denominators
// ---------------------------------------------------------------------------
__global__ void edge_expden(float* __restrict__ exbuf, const int* __restrict__ dst,
                            const unsigned int* __restrict__ nodemax,
                            float* __restrict__ den, int E)
{
    long i = (long)blockIdx.x * blockDim.x + threadIdx.x;
    if (i >= (long)E * 4) return;
    long e = i >> 2; int h = (int)(i & 3);
    int d = dst[e];
    float mx = funord(nodemax[(long)d * 4 + h]);
    float ex = expf(exbuf[i] - mx);
    exbuf[i] = ex;
    atomicAdd(&den[(long)d * 4 + h], ex);
}

// ---------------------------------------------------------------------------
// Pass 3: out[dst] += alpha * xl[src]  (wave per edge)
// ---------------------------------------------------------------------------
__global__ __launch_bounds__(256) void edge_scatter(
    const float* __restrict__ exbuf, const float* __restrict__ den,
    const float* __restrict__ xl, const int* __restrict__ src,
    const int* __restrict__ dst, float* __restrict__ out, int E)
{
    const int lane = threadIdx.x & 31;
    const int wave = threadIdx.x >> 5;
    long e = (long)blockIdx.x * 8 + wave;
    if (e >= E) return;

    const int s = src[e], d = dst[e];
    const int h = lane >> 3, f = lane * 8;
    float alpha = exbuf[e * 4 + h] / den[(long)d * 4 + h];

    const float* xs = &xl[(long)s * 256 + f];
    float*       od = &out[(long)d * 256 + f];
    float4 v0 = *(const float4*)&xs[0];
    float4 v1 = *(const float4*)&xs[4];
    atomicAdd(&od[0], alpha * v0.x); atomicAdd(&od[1], alpha * v0.y);
    atomicAdd(&od[2], alpha * v0.z); atomicAdd(&od[3], alpha * v0.w);
    atomicAdd(&od[4], alpha * v1.x); atomicAdd(&od[5], alpha * v1.y);
    atomicAdd(&od[6], alpha * v1.z); atomicAdd(&od[7], alpha * v1.w);
}

// ---------------------------------------------------------------------------
// Node update: h += gelu(layernorm(agg + cb))   (wave per node)
// ---------------------------------------------------------------------------
__global__ __launch_bounds__(256) void node_update(
    const float* __restrict__ agg, const float* __restrict__ cb,
    const float* __restrict__ gamma, const float* __restrict__ beta,
    float* __restrict__ h, int N)
{
    const int lane = threadIdx.x & 31;
    const int wave = threadIdx.x >> 5;
    long n = (long)blockIdx.x * 8 + wave;
    if (n >= N) return;
    const int f = lane * 8;

    float v[8];
    float s = 0.0f;
#pragma unroll
    for (int j = 0; j < 8; ++j) { v[j] = agg[n * 256 + f + j] + cb[f + j]; s += v[j]; }
    for (int o = 1; o < 32; o <<= 1) s += __shfl_xor(s, o, 32);
    float mu = s * (1.0f / 256.0f);

    float q = 0.0f;
#pragma unroll
    for (int j = 0; j < 8; ++j) { float dvt = v[j] - mu; q += dvt * dvt; }
    for (int o = 1; o < 32; o <<= 1) q += __shfl_xor(q, o, 32);
    float rstd = rsqrtf(q * (1.0f / 256.0f) + 1e-5f);

#pragma unroll
    for (int j = 0; j < 8; ++j) {
        float y = (v[j] - mu) * rstd * gamma[f + j] + beta[f + j];
        h[n * 256 + f + j] += gelu_exact(y);
    }
}

// ---------------------------------------------------------------------------
// Graph pooling: pooled[batch[n]] += h[n]; cnt[batch[n]] += 1  (wave per node)
// ---------------------------------------------------------------------------
__global__ __launch_bounds__(256) void pool_scatter(
    const float* __restrict__ h, const int* __restrict__ batch,
    float* __restrict__ pooled, float* __restrict__ cnt, int N)
{
    const int lane = threadIdx.x & 31;
    const int wave = threadIdx.x >> 5;
    long n = (long)blockIdx.x * 8 + wave;
    if (n >= N) return;
    const int g = batch[n];
    const int f = lane * 8;
#pragma unroll
    for (int j = 0; j < 8; ++j)
        atomicAdd(&pooled[(long)g * 256 + f + j], h[n * 256 + f + j]);
    if (lane == 0) atomicAdd(&cnt[g], 1.0f);
}

__global__ void pool_div(float* __restrict__ pooled, const float* __restrict__ cnt, int G)
{
    long i = (long)blockIdx.x * blockDim.x + threadIdx.x;
    if (i >= (long)G * 256) return;
    pooled[i] /= fmaxf(cnt[i >> 8], 1.0f);
}

// ---------------------------------------------------------------------------
// Final: z / max(||z||, 1e-12) in place (wave per row)
// ---------------------------------------------------------------------------
__global__ __launch_bounds__(256) void l2norm_rows(float* __restrict__ z, int G)
{
    const int lane = threadIdx.x & 31;
    const int wave = threadIdx.x >> 5;
    long g = (long)blockIdx.x * 8 + wave;
    if (g >= G) return;
    const int f = lane * 8;
    float v[8]; float ss = 0.0f;
#pragma unroll
    for (int j = 0; j < 8; ++j) { v[j] = z[g * 256 + f + j]; ss += v[j] * v[j]; }
    for (int o = 1; o < 32; o <<= 1) ss += __shfl_xor(ss, o, 32);
    float inv = 1.0f / fmaxf(sqrtf(ss), 1e-12f);
#pragma unroll
    for (int j = 0; j < 8; ++j) z[g * 256 + f + j] = v[j] * inv;
}

// ---------------------------------------------------------------------------
extern "C" void kernel_launch(void* const* d_in, const int* in_sizes, int n_in,
                              void* d_out, int out_size, void* d_ws, size_t ws_size,
                              hipStream_t stream)
{
    const float* x     = (const float*)d_in[0];
    const int*   eidx  = (const int*)d_in[1];
    const float* eattr = (const float*)d_in[2];
    const int*   batch = (const int*)d_in[3];
    // d_in[4] = num_graphs (device scalar; G derived from out_size instead)
    const float* Wp    = (const float*)d_in[5];
    const float* bp    = (const float*)d_in[6];
    const float* Wl    = (const float*)d_in[7];
    const float* bl    = (const float*)d_in[8];
    const float* Wr    = (const float*)d_in[9];
    const float* br    = (const float*)d_in[10];
    const float* We    = (const float*)d_in[11];
    const float* att   = (const float*)d_in[12];
    const float* cb    = (const float*)d_in[13];
    const float* gamma = (const float*)d_in[14];
    const float* beta  = (const float*)d_in[15];
    const float* W1    = (const float*)d_in[16];
    const float* b1    = (const float*)d_in[17];
    const float* W2    = (const float*)d_in[18];
    const float* b2    = (const float*)d_in[19];

    const int Dh = 256;
    const int N  = in_sizes[0] / 161;
    const int E  = in_sizes[1] / 2;
    const int G  = out_size / Dh;

    const int* src = eidx;
    const int* dst = eidx + E;

    // ---- workspace carving ----------------------------------------------
    char* ws = (char*)d_ws;
    auto carve = [&](size_t bytes) {
        char* p = ws; ws += (bytes + 255) & ~(size_t)255; return p;
    };
    float*        h       = (float*)carve((size_t)N * Dh * 4);
    float*        xl      = (float*)carve((size_t)N * Dh * 4);
    float*        xr      = (float*)carve((size_t)N * Dh * 4);
    float*        agg     = (float*)carve((size_t)N * Dh * 4);
    float*        exb     = (float*)carve((size_t)E * 4 * 4);
    unsigned int* nodemax = (unsigned int*)carve((size_t)N * 4 * 4);
    float*        den     = (float*)carve((size_t)N * 4 * 4);
    float*        pooled  = (float*)carve((size_t)G * Dh * 4);
    float*        cnt     = (float*)carve((size_t)G * 4);
    float*        z1      = (float*)carve((size_t)G * Dh * 4);

    auto gemm = [&](const float* Ap, const float* Wm, const float* bi, float* O,
                    int M, int K, int act) {
        dim3 grid(Dh / 64, (M + 63) / 64);
        if (K == 161)
            gemm_wmma<161><<<grid, 256, 0, stream>>>(Ap, Wm, bi, O, M, act);
        else
            gemm_wmma<256><<<grid, 256, 0, stream>>>(Ap, Wm, bi, O, M, act);
    };
    auto fill = [&](void* p, unsigned int v, long n) {
        fill_u32<<<(unsigned)((n + 255) / 256), 256, 0, stream>>>((unsigned int*)p, v, n);
    };

    const int edgeBlocks = (E + 7) / 8;
    const int nodeBlocks = (N + 7) / 8;

    // ---- input projection ------------------------------------------------
    gemm(x, Wp, bp, h, N, 161, 0);

    // ---- 4 GATv2 layers --------------------------------------------------
    for (int l = 0; l < 4; ++l) {
        fill(agg, 0u, (long)N * Dh);
        fill(nodemax, NEG_INF_ORD, (long)N * 4);
        fill(den, 0u, (long)N * 4);

        gemm(h, Wl + (size_t)l * Dh * Dh, bl + (size_t)l * Dh, xl, N, 256, 0);
        gemm(h, Wr + (size_t)l * Dh * Dh, br + (size_t)l * Dh, xr, N, 256, 0);

        edge_logits<<<edgeBlocks, 256, 0, stream>>>(
            xl, xr, eattr, src, dst, We + (size_t)l * 11 * Dh, att + (size_t)l * Dh,
            exb, nodemax, E);
        edge_expden<<<(unsigned)(((long)E * 4 + 255) / 256), 256, 0, stream>>>(
            exb, dst, nodemax, den, E);
        edge_scatter<<<edgeBlocks, 256, 0, stream>>>(exb, den, xl, src, dst, agg, E);

        node_update<<<nodeBlocks, 256, 0, stream>>>(
            agg, cb + (size_t)l * Dh, gamma + (size_t)l * Dh, beta + (size_t)l * Dh, h, N);
    }

    // ---- mean pooling + MLP head + L2 normalize --------------------------
    fill(pooled, 0u, (long)G * Dh);
    fill(cnt, 0u, (long)G);
    pool_scatter<<<nodeBlocks, 256, 0, stream>>>(h, batch, pooled, cnt, N);
    pool_div<<<(unsigned)(((long)G * Dh + 255) / 256), 256, 0, stream>>>(pooled, cnt, G);

    gemm(pooled, W1, b1, z1, G, 256, 1);               // gelu
    gemm(z1, W2, b2, (float*)d_out, G, 256, 0);
    l2norm_rows<<<(G + 7) / 8, 256, 0, stream>>>((float*)d_out, G);
}
